// LSTMStep_55293408968975
// MI455X (gfx1250) — matrix operand order repaired and checked by reading, not compiled, not run on hardware
//
#include <hip/hip_runtime.h>
#include <hip/hip_bf16.h>
#include <math.h>

// ---------------------------------------------------------------------------
// LSTM greedy decode, B=32, T=32, C=2048, U=1024.
//   z_t = [x_t, onehot(idx_{t-1})] @ Wk + h @ Wr + b
// - x_t@Wk[0:C] hoisted out of the loop: one big WMMA GEMM, time-major out.
// - onehot@Wk[C:2C] is an fp32 row gather (exact).
// - Recurrent GEMM h@Wr is fused with the LSTM gates: each wave owns the same
//   16-wide u-slice across all four gates (4 accumulators, shared A frag),
//   applies hard_sigmoid/tanh in registers, updates c and writes h16 —
//   no z round trip. h16 is ping-ponged to avoid RAW races across waves.
// - All f16 weights (~28 MB) are L2-resident on MI455X (192 MB L2), so the
//   32 sequential steps stream Wr/Ws from L2, not HBM.
// ---------------------------------------------------------------------------

#define B_   32
#define T_   32
#define C_   2048
#define U_   1024
#define N4U  4096   // 4*U

typedef _Float16 h8v  __attribute__((ext_vector_type(8)));
typedef _Float16 h16v __attribute__((ext_vector_type(16)));
typedef float    f8v  __attribute__((ext_vector_type(8)));

// ---- f32 (K x N row-major) -> f16 transposed (N x K row-major) -------------
__global__ void k_cvt_transpose(const float* __restrict__ in,
                                _Float16* __restrict__ out, int K, int N) {
    int idx = blockIdx.x * blockDim.x + threadIdx.x;
    if (idx >= K * N) return;
    int n = idx % N;
    int k = idx / N;
    out[(size_t)n * K + k] = (_Float16)in[idx];   // coalesced read
}

// ---- plain f32 -> f16 ------------------------------------------------------
__global__ void k_cvt(const float* __restrict__ in,
                      _Float16* __restrict__ out, int total) {
    int idx = blockIdx.x * blockDim.x + threadIdx.x;
    if (idx >= total) return;
    out[idx] = (_Float16)in[idx];
}

// ---- WMMA fragment builders ------------------------------------------------
// A fragment (16x32 f16): lane<16 holds row M=lane, K = k0+{0..15};
// lane>=16 holds row M=lane-16, K = k0+{16..31}.
__device__ __forceinline__ h16v load_afrag(const _Float16* Abase, int k0) {
    h16v a = *(const h16v*)(Abase + k0);
    return a;
}

__device__ __forceinline__ float hsig(float v) {
    return fminf(fmaxf(0.2f * v + 0.5f, 0.0f), 1.0f);
}

// ---- generic WMMA GEMM (used for the hoisted X@WkTop and for logits) -------
// One 16x16 output tile per wave; tile counts are exact multiples of 8
// waves/block so EXEC is all-ones at every v_wmma (ISA requirement).
__global__ __launch_bounds__(256)
void k_gemm(const _Float16* __restrict__ A,    // M x K row-major (f16)
            const _Float16* __restrict__ BT,   // N x K row-major (f16)
            float* __restrict__ Cout,
            int M, int K, int N,
            const float* __restrict__ biasVec,   // + biasVec[col] (may be null)
            int outRemapT)                       // 1: row=(b*T+t) -> (t*B+b)
{
    int wave = (int)((blockIdx.x * blockDim.x + threadIdx.x) >> 5);
    int lane = threadIdx.x & 31;
    int mtiles = M >> 4;
    int mt = wave % mtiles;
    int nt = wave / mtiles;

    int halfSel = lane >> 4;
    int lr      = lane & 15;
    const _Float16* Abase = A  + (size_t)((mt << 4) + lr) * K + (halfSel ? 16 : 0);
    const _Float16* Bbase = BT + (size_t)((nt << 4) + lr) * K + (halfSel ? 16 : 0);

    f8v acc = {};
    for (int k0 = 0; k0 < K; k0 += 32) {
        h16v a = load_afrag(Abase, k0);
        h16v b = *(const h16v*)(Bbase + k0);        // 32 contiguous bytes
        acc = __builtin_amdgcn_wmma_f32_16x16x32_f16(
                  false, a, false, b, (short)0, acc, false, false);
    }

    int col = (nt << 4) + lr;
#pragma unroll
    for (int r = 0; r < 8; ++r) {
        int row = (mt << 4) + r + (halfSel << 3);   // C/D layout: vgpr r -> M=r / r+8
        float v = acc[r];
        if (biasVec) v += biasVec[col];
        size_t orow;
        if (outRemapT) { int bb = row / T_, tt = row % T_; orow = (size_t)tt * B_ + bb; }
        else           orow = (size_t)row;
        Cout[orow * N + col] = v;
    }
}

// ---- fused recurrent GEMM + LSTM gates -------------------------------------
// 128 waves: mt in {0,1}, ut in 0..63. Each wave computes the 16x16 patch of
// all four gate blocks (cols ut*16.. of i/f/g/o), adds the hoisted Z0[t]
// block, the fp32 one-hot gather, then does the gate math in registers and
// writes c (fp32, in-place, uniquely owned) and h16_out (ping-pong buffer).
__global__ __launch_bounds__(256)
void k_rec_gates(const _Float16* __restrict__ Hin,   // h16 in  (B x U)
                 const _Float16* __restrict__ WrT,   // 4U x U (f16, transposed)
                 const float* __restrict__ Z0t,      // (B x 4U) hoisted block
                 const float* __restrict__ WkBot,    // (C x 4U) fp32, or null
                 const int*   __restrict__ idxPrev,  // (B,), or null
                 float* __restrict__ cState,         // (B x U) fp32
                 _Float16* __restrict__ Hout)        // h16 out (B x U)
{
    int wave = (int)((blockIdx.x * blockDim.x + threadIdx.x) >> 5);
    int lane = threadIdx.x & 31;
    int mt = wave & 1;
    int ut = wave >> 1;

    int halfSel = lane >> 4;
    int lr      = lane & 15;
    const _Float16* Abase = Hin + (size_t)((mt << 4) + lr) * U_ + (halfSel ? 16 : 0);
    size_t boff = (size_t)((ut << 4) + lr) * U_ + (halfSel ? 16 : 0);
    const _Float16* B0 = WrT + boff;                          // gate i rows
    const _Float16* B1 = WrT + boff + (size_t)U_ * U_;        // gate f rows
    const _Float16* B2 = WrT + boff + (size_t)2 * U_ * U_;    // gate g rows
    const _Float16* B3 = WrT + boff + (size_t)3 * U_ * U_;    // gate o rows

    f8v aI = {}, aF = {}, aG = {}, aO = {};
    for (int k0 = 0; k0 < U_; k0 += 32) {
        h16v a = load_afrag(Abase, k0);        // shared across the 4 WMMAs
        aI = __builtin_amdgcn_wmma_f32_16x16x32_f16(
                 false, a, false, *(const h16v*)(B0 + k0), (short)0, aI, false, false);
        aF = __builtin_amdgcn_wmma_f32_16x16x32_f16(
                 false, a, false, *(const h16v*)(B1 + k0), (short)0, aF, false, false);
        aG = __builtin_amdgcn_wmma_f32_16x16x32_f16(
                 false, a, false, *(const h16v*)(B2 + k0), (short)0, aG, false, false);
        aO = __builtin_amdgcn_wmma_f32_16x16x32_f16(
                 false, a, false, *(const h16v*)(B3 + k0), (short)0, aO, false, false);
    }

    int u = (ut << 4) + lr;
#pragma unroll
    for (int r = 0; r < 8; ++r) {
        int row = (mt << 4) + r + (halfSel << 3);
        const float* zrow = Z0t + (size_t)row * N4U;
        float zi = aI[r] + zrow[u];
        float zf = aF[r] + zrow[U_ + u];
        float zg = aG[r] + zrow[2 * U_ + u];
        float zo = aO[r] + zrow[3 * U_ + u];
        if (WkBot) {                               // one-hot feedback gather
            const float* g = WkBot + (size_t)idxPrev[row] * N4U;
            zi += g[u]; zf += g[U_ + u]; zg += g[2 * U_ + u]; zo += g[3 * U_ + u];
        }
        float I = hsig(zi), F = hsig(zf), G = tanhf(zg), O = hsig(zo);
        size_t ci = (size_t)row * U_ + u;          // uniquely owned -> no race
        float cn = F * cState[ci] + I * G;
        cState[ci] = cn;
        Hout[ci] = (_Float16)(O * tanhf(cn));
    }
}

// ---- per-row argmax (first-max tie rule, like jnp.argmax) ------------------
__global__ void k_argmax(const float* __restrict__ logits,   // (B, C)
                         int* __restrict__ idxBuf,           // (B,)
                         int* __restrict__ outIdx, int t) {  // out[b*T + t]
    __shared__ float sv[256];
    __shared__ int   si[256];
    int b = blockIdx.x;
    const float* row = logits + (size_t)b * C_;
    float best = -INFINITY; int bi = 0;
    for (int cidx = threadIdx.x; cidx < C_; cidx += blockDim.x) {
        float v = row[cidx];
        if (v > best) { best = v; bi = cidx; }   // strictly > keeps first max
    }
    sv[threadIdx.x] = best; si[threadIdx.x] = bi;
    __syncthreads();
    for (int s = blockDim.x >> 1; s > 0; s >>= 1) {
        if ((int)threadIdx.x < s) {
            float v2 = sv[threadIdx.x + s]; int i2 = si[threadIdx.x + s];
            if (v2 > sv[threadIdx.x] ||
                (v2 == sv[threadIdx.x] && i2 < si[threadIdx.x])) {
                sv[threadIdx.x] = v2; si[threadIdx.x] = i2;
            }
        }
        __syncthreads();
    }
    if (threadIdx.x == 0) { idxBuf[b] = si[0]; outIdx[b * T_ + t] = si[0]; }
}

// ---------------------------------------------------------------------------
extern "C" void kernel_launch(void* const* d_in, const int* in_sizes, int n_in,
                              void* d_out, int out_size, void* d_ws, size_t ws_size,
                              hipStream_t stream) {
    const float* x  = (const float*)d_in[0];   // (B, T, C)
    const float* Wk = (const float*)d_in[1];   // (2C, 4U)
    const float* Wr = (const float*)d_in[2];   // (U, 4U)
    const float* bb = (const float*)d_in[3];   // (4U,)
    const float* Ws = (const float*)d_in[4];   // (U, C)
    const float* bs = (const float*)d_in[5];   // (C,)
    int* out = (int*)d_out;                    // (B, T) int32 argmax indices

    char* ws = (char*)d_ws;
    size_t off = 0;
    _Float16* WkTopT = (_Float16*)(ws + off); off += (size_t)N4U * C_ * 2;      // 16 MB
    _Float16* WrT    = (_Float16*)(ws + off); off += (size_t)N4U * U_ * 2;      //  8 MB
    _Float16* WsT    = (_Float16*)(ws + off); off += (size_t)C_  * U_ * 2;      //  4 MB
    _Float16* X16    = (_Float16*)(ws + off); off += (size_t)B_ * T_ * C_ * 2;  //  4 MB
    float*    Z0     = (float*)   (ws + off); off += (size_t)T_ * B_ * N4U * 4; // 16 MB
    float*    logits = (float*)   (ws + off); off += (size_t)B_ * C_ * 4;
    float*    cState = (float*)   (ws + off); off += (size_t)B_ * U_ * 4;
    _Float16* h16a   = (_Float16*)(ws + off); off += (size_t)B_ * U_ * 2;       // ping
    _Float16* h16b   = (_Float16*)(ws + off); off += (size_t)B_ * U_ * 2;       // pong
    int*      idxBuf = (int*)     (ws + off); off += (size_t)B_ * 4;

    // --- per-call weight conversion / transposition + state init ---
    k_cvt_transpose<<<(C_ * N4U + 255) / 256, 256, 0, stream>>>(Wk, WkTopT, C_, N4U);
    k_cvt_transpose<<<(U_ * N4U + 255) / 256, 256, 0, stream>>>(Wr, WrT, U_, N4U);
    k_cvt_transpose<<<(U_ * C_  + 255) / 256, 256, 0, stream>>>(Ws, WsT, U_, C_);
    k_cvt<<<(B_ * T_ * C_ + 255) / 256, 256, 0, stream>>>(x, X16, B_ * T_ * C_);
    hipMemsetAsync(cState, 0, (size_t)B_ * U_ * 4, stream);
    hipMemsetAsync(h16a,   0, (size_t)B_ * U_ * 2, stream);

    // --- hoisted GEMM: Z0[t][b][:] = x[b][t][:] @ WkTop + b  (time-major) ---
    // M = B*T = 1024 -> 64*256 = 16384 tiles -> 2048 blocks of 8 waves.
    k_gemm<<<2048, 256, 0, stream>>>(X16, WkTopT, Z0, B_ * T_, C_, N4U, bb, 1);

    const float* WkBot = Wk + (size_t)C_ * N4U;   // rows C..2C-1, fp32 gather

    for (int t = 0; t < T_; ++t) {
        _Float16* hIn  = (t & 1) ? h16b : h16a;   // ping-pong (no RAW race)
        _Float16* hOut = (t & 1) ? h16a : h16b;
        // fused: z = Z0[t] + Wk[C+idx_prev] + h@Wr -> gates -> c, h
        // 128 waves = 16 blocks of 8 waves.
        k_rec_gates<<<16, 256, 0, stream>>>(hIn, WrT,
                                            Z0 + (size_t)t * B_ * N4U,
                                            t ? WkBot : nullptr,
                                            t ? idxBuf : nullptr,
                                            cState, hOut);
        // logits = h_new @ Ws + bs : 2*128 = 256 tiles -> 32 blocks.
        k_gemm<<<32, 256, 0, stream>>>(hOut, WsT, logits, B_, U_, C_, bs, 0);
        k_argmax<<<B_, 256, 0, stream>>>(logits, idxBuf, out, t);
    }
    (void)in_sizes; (void)n_in; (void)out_size; (void)ws_size;
}
